// IntraMetapathAggregation_86706799772308
// MI455X (gfx1250) — matrix operand order repaired and checked by reading, not compiled, not run on hardware
//
#include <hip/hip_runtime.h>

#define N_NODES 50000
#define N_EDGES 1000000
#define LPATH   3
#define DIN     64
#define KHEADS  8
#define DOUTDIM 8
#define SLOPE   0.01f
#define NTILES  (N_EDGES / 16)   // 62500 tiles of 16 edges

typedef __attribute__((ext_vector_type(16))) __bf16 v16bf;
typedef __attribute__((ext_vector_type(8)))  float  v8f;
typedef __attribute__((ext_vector_type(4)))  float  f4;   // native vector for NT ld/st

union BFrag { v16bf v; uint4 q[2]; };

__device__ __forceinline__ unsigned short f2bf(float f) {
  unsigned u = __float_as_uint(f);
  u += 0x7FFFu + ((u >> 16) & 1u);          // round-to-nearest-even
  return (unsigned short)(u >> 16);
}
__device__ __forceinline__ unsigned mapf(float f) {   // order-preserving f32 -> u32
  unsigned u = __float_as_uint(f);
  return (u & 0x80000000u) ? ~u : (u | 0x80000000u);
}
__device__ __forceinline__ float unmapf(unsigned u) {
  unsigned v = (u & 0x80000000u) ? (u & 0x7FFFFFFFu) : ~u;
  return __uint_as_float(v);
}

// Pass 1: er = node_feat @ W_r^T ; init out / segment-max / segment-sum
__global__ void hgnn_pass1(const float* __restrict__ node_feat,
                           const float* __restrict__ W_r,
                           float* __restrict__ er,
                           unsigned* __restrict__ mmax,
                           float* __restrict__ sden,
                           float* __restrict__ out) {
  int i = blockIdx.x * blockDim.x + threadIdx.x;
  if (i < N_NODES * 64) out[i] = 0.0f;
  if (i < N_NODES * KHEADS) {
    int node = i >> 3, k = i & 7;
    const f4* nf = (const f4*)(node_feat + node * DIN);
    const f4* wr = (const f4*)(W_r + k * DIN);
    float acc = 0.f;
#pragma unroll
    for (int j = 0; j < 16; ++j) {
      f4 a = nf[j], b = wr[j];
      acc += a.x * b.x + a.y * b.y + a.z * b.z + a.w * b.w;
    }
    er[i]   = acc;
    mmax[i] = 0u;     // unmaps to "no edge seen"; only read for nodes with edges
    sden[i] = 0.f;
  }
}

// Pass 2: per 16-edge tile: mean -> bf16, WMMA encoder GEMM (16x64x64),
// store ef, compute attention logits e, atomic segment-max.
__global__ void __launch_bounds__(256)
hgnn_pass2(const float* __restrict__ edge_feat, const int* __restrict__ dst,
           const float* __restrict__ W_enc, const float* __restrict__ attn_l,
           const float* __restrict__ er,
           float* __restrict__ ef_ws, float* __restrict__ e_ws,
           unsigned* __restrict__ mmax) {
  __shared__ __align__(16) unsigned short Wt[64 * 72];        // W_enc^T, bf16, padded rows
  __shared__ __align__(16) unsigned short Am[8][16 * 72];     // per-wave mean tile, bf16
  __shared__ __align__(16) float          eftile[8][16 * 68]; // per-wave ef tile, f32
  __shared__ float attnl_s[64];

  const int tid  = threadIdx.x;
  const int wave = tid >> 5;
  const int lane = tid & 31;

  // Stage W_enc^T as bf16: Wt[n][k] = W_enc[k][n]
  for (int idx = tid; idx < 64 * 64; idx += 256) {
    int n = idx & 63, k = idx >> 6;
    Wt[n * 72 + k] = f2bf(W_enc[k * 64 + n]);
  }
  if (tid < 64) attnl_s[tid] = attn_l[tid];
  __syncthreads();

  const int  tile  = blockIdx.x * 8 + wave;
  const bool valid = tile < NTILES;
  const int  sbase = valid ? tile * 16 : 0;   // clamp: tail waves do safe dummy work

  // Mean over L=3, convert to bf16 into Am[wave] (streaming: non-temporal loads)
  for (int i = lane; i < 256; i += 32) {        // 16 rows x 16 float4-cols
    int m = i >> 4, c4 = i & 15;
    const f4* p =
        (const f4*)(edge_feat + (size_t)(sbase + m) * (LPATH * DIN)) + c4;
    f4 a = __builtin_nontemporal_load(p);
    f4 b = __builtin_nontemporal_load(p + 16);
    f4 c = __builtin_nontemporal_load(p + 32);
    float s0 = (a.x + b.x + c.x) * (1.f / 3.f);
    float s1 = (a.y + b.y + c.y) * (1.f / 3.f);
    float s2 = (a.z + b.z + c.z) * (1.f / 3.f);
    float s3 = (a.w + b.w + c.w) * (1.f / 3.f);
    unsigned* q32 = (unsigned*)&Am[wave][m * 72 + c4 * 4];
    q32[0] = (unsigned)f2bf(s0) | ((unsigned)f2bf(s1) << 16);
    q32[1] = (unsigned)f2bf(s2) | ((unsigned)f2bf(s3) << 16);
  }
  __syncthreads();

  const int n  = lane & 15;
  const int hi = lane >> 4;

  // WMMA: D(16x64) = A(16x64) * W_enc(64x64), bf16 in, f32 accum.
  // A frag (16x32 bf16): lane m=n; elems 0..7 -> K0+hi*8+0..7, 8..15 -> K0+16+hi*8+0..7
  // B frag (32x16 bf16): lane col=n; elems 0..7 -> K0+hi*8+0..7, 8..15 -> K0+16+hi*8+0..7
  v8f C[4] = {};
  const unsigned short* arow = &Am[wave][n * 72];
#pragma unroll
  for (int c = 0; c < 2; ++c) {
    const int K0 = c * 32;
    BFrag af;
    af.q[0] = *(const uint4*)(arow + K0 + hi * 8);
    af.q[1] = *(const uint4*)(arow + K0 + hi * 8 + 16);
#pragma unroll
    for (int t = 0; t < 4; ++t) {
      BFrag bf;
      const unsigned short* brow = &Wt[(t * 16 + n) * 72 + K0 + hi * 8];
      bf.q[0] = *(const uint4*)(brow);
      bf.q[1] = *(const uint4*)(brow + 16);
      C[t] = __builtin_amdgcn_wmma_f32_16x16x32_bf16(
          false, af.v, false, bf.v, (short)0, C[t], false, false);
    }
  }

  // Stage C tile to LDS. Layout: lanes 0-15 -> M=r, N=lane; 16-31 -> M=8+r, N=lane-16
#pragma unroll
  for (int t = 0; t < 4; ++t) {
#pragma unroll
    for (int r = 0; r < 8; ++r) {
      eftile[wave][(hi * 8 + r) * 68 + t * 16 + n] = C[t][r];
    }
  }
  __syncthreads();

  // Coalesced ef tile writeback from LDS (one branch, b128 stores, non-temporal)
  if (valid) {
    for (int i = lane; i < 256; i += 32) {      // 16 rows x 16 float4-cols
      int r = i >> 4, c4 = i & 15;
      f4 v = *(const f4*)&eftile[wave][r * 68 + c4 * 4];
      __builtin_nontemporal_store(
          v, (f4*)(ef_ws + (size_t)(sbase + r) * 64) + c4);
    }
  }

  // Attention logits: each lane handles edge m = lane&15, 4 heads (hi picks half)
  const int m     = lane & 15;
  const int eidx  = sbase + m;
  const int dnode = dst[eidx];
  const float* efr = &eftile[wave][m * 68];
#pragma unroll
  for (int kk = 0; kk < 4; ++kk) {
    const int k = hi * 4 + kk;
    f4 x0 = *(const f4*)(efr + k * 8);
    f4 x1 = *(const f4*)(efr + k * 8 + 4);
    f4 a0 = *(const f4*)(attnl_s + k * 8);
    f4 a1 = *(const f4*)(attnl_s + k * 8 + 4);
    float acc = x0.x * a0.x + x0.y * a0.y + x0.z * a0.z + x0.w * a0.w +
                x1.x * a1.x + x1.y * a1.y + x1.z * a1.z + x1.w * a1.w;
    float ev = acc + er[dnode * 8 + k];
    ev = ev > 0.f ? ev : SLOPE * ev;          // LeakyReLU
    if (valid) {
      e_ws[(size_t)eidx * 8 + k] = ev;
      atomicMax(&mmax[dnode * 8 + k], mapf(ev));
    }
  }
}

// Pass 3: ex = exp(e - m[dst]) in place; segment-sum denominators
__global__ void hgnn_pass3(const int* __restrict__ dst,
                           const unsigned* __restrict__ mmax,
                           float* __restrict__ e_ws,
                           float* __restrict__ sden) {
  int i = blockIdx.x * blockDim.x + threadIdx.x;
  if (i >= N_EDGES * KHEADS) return;
  int edge = i >> 3, k = i & 7;
  int d = dst[edge];
  float x = __expf(e_ws[i] - unmapf(mmax[d * 8 + k]));
  e_ws[i] = x;
  atomicAdd(&sden[d * 8 + k], x);
}

// Pass 4: out[dst] += ef * (ex / s[dst])
__global__ void hgnn_pass4(const int* __restrict__ dst,
                           const float* __restrict__ e_ws,
                           const float* __restrict__ sden,
                           const float* __restrict__ ef_ws,
                           float* __restrict__ out) {
  int i = blockIdx.x * blockDim.x + threadIdx.x;
  if (i >= N_EDGES * KHEADS) return;
  int edge = i >> 3, k = i & 7;
  int d = dst[edge];
  float a = e_ws[i] / sden[d * 8 + k];
  const f4* efp = (const f4*)(ef_ws + (size_t)edge * 64 + k * 8);
  f4 u0 = __builtin_nontemporal_load(efp);
  f4 u1 = __builtin_nontemporal_load(efp + 1);
  float* op = out + (size_t)d * 64 + k * 8;
  atomicAdd(op + 0, u0.x * a);
  atomicAdd(op + 1, u0.y * a);
  atomicAdd(op + 2, u0.z * a);
  atomicAdd(op + 3, u0.w * a);
  atomicAdd(op + 4, u1.x * a);
  atomicAdd(op + 5, u1.y * a);
  atomicAdd(op + 6, u1.z * a);
  atomicAdd(op + 7, u1.w * a);
}

extern "C" void kernel_launch(void* const* d_in, const int* in_sizes, int n_in,
                              void* d_out, int out_size, void* d_ws, size_t ws_size,
                              hipStream_t stream) {
  (void)in_sizes; (void)n_in; (void)out_size; (void)ws_size;
  const float* node_feat = (const float*)d_in[0];
  const float* edge_feat = (const float*)d_in[1];
  const int*   dst       = (const int*)d_in[2];
  const float* W_enc     = (const float*)d_in[3];
  const float* attn_l    = (const float*)d_in[4];
  const float* W_r       = (const float*)d_in[5];
  float* out = (float*)d_out;

  char* w = (char*)d_ws;
  float*    ef_ws = (float*)(w);                       // E*64 f32 = 256,000,000 B
  float*    e_ws  = (float*)(w + 256000000ull);        // E*8  f32 =  32,000,000 B
  float*    er    = (float*)(w + 288000000ull);        // N*8  f32 =   1,600,000 B
  unsigned* mmax  = (unsigned*)(w + 289600000ull);     // N*8  u32 =   1,600,000 B
  float*    sden  = (float*)(w + 291200000ull);        // N*8  f32 =   1,600,000 B

  hgnn_pass1<<<(N_NODES * 64 + 255) / 256, 256, 0, stream>>>(
      node_feat, W_r, er, mmax, sden, out);
  hgnn_pass2<<<(NTILES + 7) / 8, 256, 0, stream>>>(
      edge_feat, dst, W_enc, attn_l, er, ef_ws, e_ws, mmax);
  hgnn_pass3<<<(N_EDGES * KHEADS + 255) / 256, 256, 0, stream>>>(
      dst, mmax, e_ws, sden);
  hgnn_pass4<<<(N_EDGES * KHEADS + 255) / 256, 256, 0, stream>>>(
      dst, e_ws, sden, ef_ws, out);
}
